// Net_90744069030481
// MI455X (gfx1250) — compile-verified, matrix operand order checked by reading, not run on hardware
//
#include <hip/hip_runtime.h>
#include <hip/hip_bf16.h>
#include <math.h>

#define N_NODES 50000
#define N_EDGES 640000
#define NHID    128
#define HEAD    4
#define DH      32
#define G_BATCH 256
#define C_CLS   10

typedef __attribute__((ext_vector_type(16))) __bf16 v16bf;
typedef __attribute__((ext_vector_type(8)))  __bf16 v8bf;
typedef __attribute__((ext_vector_type(4)))  __bf16 v4bf;
typedef __attribute__((ext_vector_type(8)))  float  v8f;

// ---------------- utility ----------------
__global__ void fill_u32(unsigned* __restrict__ p, unsigned v, int n) {
    int i = blockIdx.x * blockDim.x + threadIdx.x;
    if (i < n) p[i] = v;
}

// transpose a batch of 128x128 matrices: Wt[m][n][k] = W[m][k][n]
__global__ void transpose128(const float* __restrict__ W, float* __restrict__ Wt, int total) {
    int i = blockIdx.x * blockDim.x + threadIdx.x;
    if (i >= total) return;
    int mat = i >> 14;
    int within = i & 16383;
    int k = within >> 7, n = within & 127;
    Wt[(size_t)mat * 16384 + n * 128 + k] = W[(size_t)mat * 16384 + k * 128 + n];
}

// monotonic float<->uint encoding so atomicMax(uint) == float max (handles negatives)
__device__ __forceinline__ unsigned fenc(float x) {
    unsigned u = __float_as_uint(x);
    return (u & 0x80000000u) ? ~u : (u | 0x80000000u);
}
__device__ __forceinline__ float fdec(unsigned u) {
    return (u & 0x80000000u) ? __uint_as_float(u & 0x7fffffffu) : __uint_as_float(~u);
}

// load 16 bf16 fragment halves: 8 at p[0..7], 8 at p[16..23] (ISA 16-bit frag K layout)
__device__ __forceinline__ v16bf load_frag(const __bf16* p) {
    v8bf lo = *(const v8bf*)(p);
    v8bf hi = *(const v8bf*)(p + 16);
    return __builtin_shufflevector(lo, hi, 0,1,2,3,4,5,6,7,8,9,10,11,12,13,14,15);
}

// ---------------- bf16 WMMA GEMM: Out[nrows x 128] = A[nrows x 128] @ W[128 x 128] (+bias)
// Wt is the PRE-TRANSPOSED weight (N-major: Wt[n*128+k]) ----------------
__global__ __launch_bounds__(256) void gemm128_wmma(
    const float* __restrict__ A, const float* __restrict__ Wt,
    const float* __restrict__ bias, float* __restrict__ Out,
    int nrows, int add_bias)
{
    __shared__ __align__(16) char smem[48 * 1024];
    __bf16* sA  = (__bf16*)smem;              // 64 x 128 bf16 = 16 KB
    __bf16* sBt = sA + 64 * 128;              // 128 x 128 bf16 = 32 KB (N-major)
    float*  sD  = (float*)smem;               // 64 x 128 f32 = 32 KB (reused after compute)

    const int tid  = threadIdx.x;
    const int row0 = blockIdx.x * 64;

    // stage transposed weights (vectorized: float4 load -> 4x bf16 -> b64 LDS store)
    for (int i = tid; i < 128 * 128 / 4; i += 256) {
        float4 w = ((const float4*)Wt)[i];
        v4bf b; b[0] = (__bf16)w.x; b[1] = (__bf16)w.y; b[2] = (__bf16)w.z; b[3] = (__bf16)w.w;
        *(v4bf*)(sBt + i * 4) = b;
    }
    // stage A tile (64 x 128), zero-pad past nrows
    for (int i = tid; i < 64 * 128 / 4; i += 256) {
        int e0 = i * 4;
        int r = e0 >> 7, c = e0 & 127;
        int gr = row0 + r;
        v4bf b;
        if (gr < nrows) {
            float4 w = *(const float4*)(A + (size_t)gr * 128 + c);
            b[0] = (__bf16)w.x; b[1] = (__bf16)w.y; b[2] = (__bf16)w.z; b[3] = (__bf16)w.w;
        } else {
            b[0] = b[1] = b[2] = b[3] = (__bf16)0.0f;
        }
        *(v4bf*)(sA + e0) = b;
    }
    __syncthreads();

    const int wave  = tid >> 5;
    const int lane  = tid & 31;
    const int ncol0 = wave * 16;          // each wave owns one 16-col tile
    const int lrow  = lane & 15;
    const int khi   = (lane >= 16) ? 8 : 0;

    v8f acc[4];
#pragma unroll
    for (int m = 0; m < 4; ++m)
#pragma unroll
        for (int r = 0; r < 8; ++r) acc[m][r] = 0.0f;

#pragma unroll
    for (int kt = 0; kt < 4; ++kt) {      // K = 128 in chunks of 32
        const int kbase = kt * 32 + khi;
        v16bf bfrag = load_frag(&sBt[(ncol0 + lrow) * 128 + kbase]);
#pragma unroll
        for (int m = 0; m < 4; ++m) {
            v16bf afrag = load_frag(&sA[(m * 16 + lrow) * 128 + kbase]);
            acc[m] = __builtin_amdgcn_wmma_f32_16x16x32_bf16(
                false, afrag, false, bfrag, (short)0, acc[m], false, false);
        }
    }

    __syncthreads();   // done reading sA/sBt; reuse smem for the output tile
    // D layout: VGPR r -> M = r (+8 for lanes>=16), N = lane%16
#pragma unroll
    for (int m = 0; m < 4; ++m)
#pragma unroll
        for (int r = 0; r < 8; ++r)
            sD[(m * 16 + r + khi) * 128 + ncol0 + lrow] = acc[m][r];
    __syncthreads();

    // coalesced write-out with fused bias (b128 stores)
    for (int i = tid; i < 64 * 128 / 4; i += 256) {
        int e0 = i * 4;
        int r = e0 >> 7, c = e0 & 127;
        int gr = row0 + r;
        if (gr < nrows) {
            float4 v = ((const float4*)sD)[i];
            if (add_bias) {
                float4 bb = *(const float4*)(bias + c);
                v.x += bb.x; v.y += bb.y; v.z += bb.z; v.w += bb.w;
            }
            *(float4*)(Out + (size_t)gr * 128 + c) = v;
        }
    }
}

// ---------------- attention projections: a_s[n,h] = <wx[n,h,:], asrc[h,:]> ----------------
__global__ void attn_proj(const float* __restrict__ wx,
                          const float* __restrict__ asrc, const float* __restrict__ adst,
                          float* __restrict__ a_s, float* __restrict__ a_d, int n)
{
    int i = blockIdx.x * blockDim.x + threadIdx.x;
    if (i >= n * HEAD) return;
    int node = i >> 2, h = i & 3;
    const float* row = wx + (size_t)node * NHID + h * DH;
    const float* vs  = asrc + h * DH;
    const float* vd  = adst + h * DH;
    float ss = 0.f, sd = 0.f;
#pragma unroll
    for (int d = 0; d < DH; ++d) { float x = row[d]; ss += x * vs[d]; sd += x * vd[d]; }
    a_s[i] = ss; a_d[i] = sd;
}

// ---------------- edge pass 1: score + segment max over dst ----------------
__global__ void edge_score_max(const int* __restrict__ ei, const float* __restrict__ ea,
                               const float* __restrict__ a_s, const float* __restrict__ a_d,
                               float* __restrict__ eval, unsigned* __restrict__ emax)
{
    int i = blockIdx.x * blockDim.x + threadIdx.x;
    if (i >= N_EDGES * HEAD) return;
    int e = i >> 2, h = i & 3;
    int s = ei[e], d = ei[N_EDGES + e];
    float v = a_s[s * HEAD + h] + a_d[d * HEAD + h];
    v = (v > 0.f) ? v : 0.2f * v;            // leaky_relu(., 0.2)
    v *= ea[e];
    eval[i] = v;
    atomicMax(&emax[d * HEAD + h], fenc(v));
}

// ---------------- edge pass 2: exp + segment sum ----------------
__global__ void edge_exp_sum(const int* __restrict__ ei, float* __restrict__ eval,
                             const unsigned* __restrict__ emax, float* __restrict__ denom)
{
    int i = blockIdx.x * blockDim.x + threadIdx.x;
    if (i >= N_EDGES * HEAD) return;
    int e = i >> 2, h = i & 3;
    int d = ei[N_EDGES + e];
    float m  = fdec(emax[d * HEAD + h]);
    float ex = __expf(eval[i] - m);
    eval[i] = ex;
    atomicAdd(&denom[d * HEAD + h], ex);
}

// ---------------- edge pass 3: alpha * wx[src] scattered into agg[dst] ----------------
__global__ __launch_bounds__(128) void edge_message(
    const int* __restrict__ ei, const float* __restrict__ eval,
    const float* __restrict__ denom, const float* __restrict__ wx,
    float* __restrict__ agg)
{
    int e   = blockIdx.x;
    int dft = threadIdx.x;          // feature index 0..127
    int h   = dft >> 5;
    int s = ei[e], d = ei[N_EDGES + e];
    float alpha = eval[e * HEAD + h] / (denom[d * HEAD + h] + 1e-16f);
    atomicAdd(&agg[(size_t)d * NHID + dft], alpha * wx[(size_t)s * NHID + dft]);
}

// ---------------- elementwise ----------------
__global__ void elu_k(const float* __restrict__ a, float* __restrict__ t, int n) {
    int i = blockIdx.x * blockDim.x + threadIdx.x;
    if (i < n) { float x = a[i]; t[i] = (x > 0.f) ? x : (__expf(x) - 1.f); }
}
__global__ void add_k(float* __restrict__ h, const float* __restrict__ t, int n) {
    int i = blockIdx.x * blockDim.x + threadIdx.x;
    if (i < n) h[i] += t[i];
}

// ---------------- global add pool ----------------
__global__ __launch_bounds__(128) void pool_k(const float* __restrict__ h,
                                              const int* __restrict__ batch,
                                              float* __restrict__ pooled)
{
    int n = blockIdx.x, dft = threadIdx.x;
    atomicAdd(&pooled[batch[n] * NHID + dft], h[(size_t)n * NHID + dft]);
}

// ---------------- tiny MLP head ----------------
__global__ void small_gemm(const float* __restrict__ A, const float* __restrict__ W,
                           const float* __restrict__ b, float* __restrict__ Out,
                           int rows, int K, int NC)
{
    int i = blockIdx.x * blockDim.x + threadIdx.x;
    if (i >= rows * NC) return;
    int r = i / NC, c = i % NC;
    float acc = b[c];
    for (int k = 0; k < K; ++k) acc += A[r * K + k] * W[k * NC + c];
    Out[i] = acc;
}

__global__ __launch_bounds__(256) void bn_k(const float* __restrict__ Z,
                                            const float* __restrict__ g, const float* __restrict__ b,
                                            float* __restrict__ Out, int NC, int do_relu)
{
    __shared__ float s1[256], s2[256];
    int c = blockIdx.x, r = threadIdx.x;     // G_BATCH == 256 rows
    float z = Z[r * NC + c];
    s1[r] = z; s2[r] = z * z;
    __syncthreads();
    for (int st = 128; st > 0; st >>= 1) {
        if (r < st) { s1[r] += s1[r + st]; s2[r] += s2[r + st]; }
        __syncthreads();
    }
    float m   = s1[0] * (1.f / G_BATCH);
    float var = s2[0] * (1.f / G_BATCH) - m * m;
    float y = (z - m) * rsqrtf(var + 1e-5f) * g[c] + b[c];
    if (do_relu) y = fmaxf(y, 0.f);
    Out[r * NC + c] = y;
}

// ---------------- driver ----------------
extern "C" void kernel_launch(void* const* d_in, const int* in_sizes, int n_in,
                              void* d_out, int out_size, void* d_ws, size_t ws_size,
                              hipStream_t stream)
{
    const float* x         = (const float*)d_in[0];
    const float* edge_attr = (const float*)d_in[1];
    const float* lin_w     = (const float*)d_in[2];
    const float* lin_b     = (const float*)d_in[3];
    const float* conv_W    = (const float*)d_in[4];   // [3,2,128,128]
    const float* conv_asrc = (const float*)d_in[5];   // [3,2,4,32]
    const float* conv_adst = (const float*)d_in[6];
    const float* fc_w1 = (const float*)d_in[7];  const float* fc_b1 = (const float*)d_in[8];
    const float* bn_g1 = (const float*)d_in[9];  const float* bn_b1 = (const float*)d_in[10];
    const float* fc_w2 = (const float*)d_in[11]; const float* fc_b2 = (const float*)d_in[12];
    const float* bn_g2 = (const float*)d_in[13]; const float* bn_b2 = (const float*)d_in[14];
    const float* fc_w3 = (const float*)d_in[15]; const float* fc_b3 = (const float*)d_in[16];
    const float* bn_g3 = (const float*)d_in[17]; const float* bn_b3 = (const float*)d_in[18];
    const int*   ei    = (const int*)d_in[19];        // [2,E]
    const int*   batch = (const int*)d_in[20];

    // workspace carving (floats)
    float* h     = (float*)d_ws;                 // N x 128
    float* t     = h     + (size_t)N_NODES * NHID;
    float* wx    = t     + (size_t)N_NODES * NHID;
    float* agg   = wx    + (size_t)N_NODES * NHID;
    float* a_s   = agg   + (size_t)N_NODES * NHID;
    float* a_d   = a_s   + (size_t)N_NODES * HEAD;
    float* denom = a_d   + (size_t)N_NODES * HEAD;
    unsigned* emax = (unsigned*)(denom + (size_t)N_NODES * HEAD);
    float* eval  = (float*)(emax + (size_t)N_NODES * HEAD);
    float* pooled= eval  + (size_t)N_EDGES * HEAD;
    float* z1    = pooled + (size_t)G_BATCH * NHID;
    float* z1n   = z1   + G_BATCH * 64;
    float* z2    = z1n  + G_BATCH * 64;
    float* z2n   = z2   + G_BATCH * 32;
    float* z3    = z2n  + G_BATCH * 32;
    float* WT    = z3   + G_BATCH * C_CLS;       // 7 x 128 x 128 pre-transposed weights

    const int gemm_blocks = (N_NODES + 63) / 64;
    const int nh  = N_NODES * HEAD;
    const int nf  = N_NODES * NHID;
    const int ehd = N_EDGES * HEAD;

    // pre-transpose all 128x128 weight matrices: WT[0]=lin_w^T, WT[1..6]=conv_W^T
    transpose128<<<(16384 + 255) / 256, 256, 0, stream>>>(lin_w, WT, 16384);
    transpose128<<<(6 * 16384 + 255) / 256, 256, 0, stream>>>(conv_W, WT + 16384, 6 * 16384);

    // h = x @ lin_w + lin_b
    gemm128_wmma<<<gemm_blocks, 256, 0, stream>>>(x, WT, lin_b, h, N_NODES, 1);

    for (int i = 0; i < 3; ++i) {
        const float* cur = h;
        for (int l = 0; l < 2; ++l) {
            const int idx = i * 2 + l;
            const float* Wlt = WT + (size_t)(1 + idx) * 16384;
            const float* as  = conv_asrc + (size_t)idx * HEAD * DH;
            const float* ad  = conv_adst + (size_t)idx * HEAD * DH;

            gemm128_wmma<<<gemm_blocks, 256, 0, stream>>>(cur, Wlt, lin_b, wx, N_NODES, 0);
            attn_proj<<<(nh + 255) / 256, 256, 0, stream>>>(wx, as, ad, a_s, a_d, N_NODES);

            fill_u32<<<(nh + 255) / 256, 256, 0, stream>>>(emax, 0u, nh);          // encoded -inf floor
            fill_u32<<<(nh + 255) / 256, 256, 0, stream>>>((unsigned*)denom, 0u, nh);
            fill_u32<<<(nf + 255) / 256, 256, 0, stream>>>((unsigned*)agg, 0u, nf);

            edge_score_max<<<(ehd + 255) / 256, 256, 0, stream>>>(ei, edge_attr, a_s, a_d, eval, emax);
            edge_exp_sum  <<<(ehd + 255) / 256, 256, 0, stream>>>(ei, eval, emax, denom);
            edge_message  <<<N_EDGES, 128, 0, stream>>>(ei, eval, denom, wx, agg);

            elu_k<<<(nf + 255) / 256, 256, 0, stream>>>(agg, t, nf);
            cur = t;
        }
        add_k<<<(nf + 255) / 256, 256, 0, stream>>>(h, t, nf);   // residual
    }

    // global add pool
    fill_u32<<<(G_BATCH * NHID + 255) / 256, 256, 0, stream>>>((unsigned*)pooled, 0u, G_BATCH * NHID);
    pool_k<<<N_NODES, 128, 0, stream>>>(h, batch, pooled);

    // MLP head with batchnorm
    small_gemm<<<(G_BATCH * 64 + 255) / 256, 256, 0, stream>>>(pooled, fc_w1, fc_b1, z1, G_BATCH, 128, 64);
    bn_k<<<64, 256, 0, stream>>>(z1, bn_g1, bn_b1, z1n, 64, 1);
    small_gemm<<<(G_BATCH * 32 + 255) / 256, 256, 0, stream>>>(z1n, fc_w2, fc_b2, z2, G_BATCH, 64, 32);
    bn_k<<<32, 256, 0, stream>>>(z2, bn_g2, bn_b2, z2n, 32, 1);
    small_gemm<<<(G_BATCH * C_CLS + 255) / 256, 256, 0, stream>>>(z2n, fc_w3, fc_b3, z3, G_BATCH, 32, C_CLS);
    bn_k<<<C_CLS, 256, 0, stream>>>(z3, bn_g3, bn_b3, (float*)d_out, C_CLS, 0);
}
